// RepresentationModel_14611478741431
// MI455X (gfx1250) — compile-verified, matrix operand order checked by reading, not run on hardware
//
#include <hip/hip_runtime.h>
#include <hip/hip_bf16.h>

// ---------------- problem constants ----------------
static constexpr int Bn  = 256;
static constexpr int Tn  = 49;
static constexpr int BEL = 2048;
static constexpr int ST  = 256;
static constexpr int ACT = 64;
static constexpr int HID = 2048;
static constexpr int EMB = 1024;

typedef __attribute__((ext_vector_type(16))) __bf16 v16bf;
typedef __attribute__((ext_vector_type(8)))  float  v8f;

__device__ __forceinline__ unsigned short f32_to_bf16(float f) {
    unsigned int u = __float_as_uint(f);
    unsigned int r = (u + 0x7FFFu + ((u >> 16) & 1u)) >> 16;   // RNE
    return (unsigned short)r;
}

__device__ __forceinline__ float sigmoidf_(float x) {
    return 1.0f / (1.0f + __expf(-x));
}

// ---------------------------------------------------------------------------
// Repack f32 weight W[K,N] (row-major, K = contraction dim) into bf16 WMMA
// B-fragment tiles. Tile = 32(K) x 16(N). Packed layout:
//   Wp[(nTile*KT + kt)*512 + lane*16 + j]
// where lane holds column (lane&15), K-half (lane>>4)*16, j = K within half.
// Each lane's 16 bf16 (32B) are contiguous -> two b128 loads in the GEMM.
// ---------------------------------------------------------------------------
__global__ void pack_w_kernel(const float* __restrict__ W,
                              unsigned short* __restrict__ Wp,
                              int K, int N) {
    size_t p = (size_t)blockIdx.x * blockDim.x + threadIdx.x;
    size_t total = (size_t)K * N;
    if (p >= total) return;
    int KT = K >> 5;
    size_t tile = p >> 9;
    int within = (int)(p & 511);
    int lane = within >> 4;
    int j    = within & 15;
    int nTile = (int)(tile / KT);
    int kt    = (int)(tile % KT);
    int col = nTile * 16 + (lane & 15);
    int k   = kt * 32 + (lane >> 4) * 16 + j;
    Wp[p] = f32_to_bf16(W[(size_t)k * N + col]);
}

// ---------------------------------------------------------------------------
// bf16 WMMA GEMM: C[M=256, N] = A[256, K] @ W[K, N] (+bias) (+relu)
// A: bf16 row-major, lda = K. W: packed fragments (above).
// 4 waves / block; each wave computes 32(M) x 64(N): 2 A-frags x 4 B-frags
// = 8 independent WMMAs per K-step. Register double-buffering pipelines the
// kt+1 fragment loads under the kt WMMAs. Block tiles 32 x 256.
// grid = (256/32, N/256).
// ---------------------------------------------------------------------------
__global__ __launch_bounds__(128)
void gemm_wmma_bf16(const unsigned short* __restrict__ A,
                    const unsigned short* __restrict__ Bp,
                    const float* __restrict__ bias,
                    float* __restrict__ outF,
                    unsigned short* __restrict__ outB,
                    int K, int N, int relu) {
    const int lane  = threadIdx.x & 31;
    const int wave  = threadIdx.x >> 5;    // 0..3
    const int lrow  = lane & 15;
    const int hsel  = lane >> 4;

    const int mBase = blockIdx.x * 32;                 // wave covers all 32 M rows
    const int nBase = blockIdx.y * 256 + wave * 64;
    const int KT    = K >> 5;
    const size_t tileStride = (size_t)KT * 512;

    const unsigned short* aPtr0 = A + (size_t)(mBase + lrow) * K + hsel * 16;
    const unsigned short* aPtr1 = aPtr0 + (size_t)16 * K;   // M rows 16..31
    const unsigned short* bPtr  = Bp + (size_t)(nBase >> 4) * tileStride + lane * 16;

    v8f acc[8] = {};   // [0..3]: M-half 0, nTiles 0..3; [4..7]: M-half 1

    // preload kt = 0
    v16bf a0c = *(const v16bf*)(aPtr0);
    v16bf a1c = *(const v16bf*)(aPtr1);
    v16bf b0c = *(const v16bf*)(bPtr + 0 * tileStride);
    v16bf b1c = *(const v16bf*)(bPtr + 1 * tileStride);
    v16bf b2c = *(const v16bf*)(bPtr + 2 * tileStride);
    v16bf b3c = *(const v16bf*)(bPtr + 3 * tileStride);

    for (int kt = 0; kt < KT - 1; ++kt) {
        // ---- issue loads for kt+1 (pipelined under the WMMAs below) ----
        const size_t ka = (size_t)(kt + 1) * 32;
        const size_t kb = (size_t)(kt + 1) * 512;
        v16bf a0n = *(const v16bf*)(aPtr0 + ka);
        v16bf a1n = *(const v16bf*)(aPtr1 + ka);
        v16bf b0n = *(const v16bf*)(bPtr + 0 * tileStride + kb);
        v16bf b1n = *(const v16bf*)(bPtr + 1 * tileStride + kb);
        v16bf b2n = *(const v16bf*)(bPtr + 2 * tileStride + kb);
        v16bf b3n = *(const v16bf*)(bPtr + 3 * tileStride + kb);
        if (kt + 2 < KT)
            __builtin_prefetch((const void*)(bPtr + (size_t)(kt + 2) * 512), 0, 1);

        // ---- 8 independent WMMAs on current fragments ----
        acc[0] = __builtin_amdgcn_wmma_f32_16x16x32_bf16(false, a0c, false, b0c, (short)0, acc[0], false, false);
        acc[4] = __builtin_amdgcn_wmma_f32_16x16x32_bf16(false, a1c, false, b0c, (short)0, acc[4], false, false);
        acc[1] = __builtin_amdgcn_wmma_f32_16x16x32_bf16(false, a0c, false, b1c, (short)0, acc[1], false, false);
        acc[5] = __builtin_amdgcn_wmma_f32_16x16x32_bf16(false, a1c, false, b1c, (short)0, acc[5], false, false);
        acc[2] = __builtin_amdgcn_wmma_f32_16x16x32_bf16(false, a0c, false, b2c, (short)0, acc[2], false, false);
        acc[6] = __builtin_amdgcn_wmma_f32_16x16x32_bf16(false, a1c, false, b2c, (short)0, acc[6], false, false);
        acc[3] = __builtin_amdgcn_wmma_f32_16x16x32_bf16(false, a0c, false, b3c, (short)0, acc[3], false, false);
        acc[7] = __builtin_amdgcn_wmma_f32_16x16x32_bf16(false, a1c, false, b3c, (short)0, acc[7], false, false);

        a0c = a0n; a1c = a1n;
        b0c = b0n; b1c = b1n; b2c = b2n; b3c = b3n;
    }

    // final K-step
    acc[0] = __builtin_amdgcn_wmma_f32_16x16x32_bf16(false, a0c, false, b0c, (short)0, acc[0], false, false);
    acc[4] = __builtin_amdgcn_wmma_f32_16x16x32_bf16(false, a1c, false, b0c, (short)0, acc[4], false, false);
    acc[1] = __builtin_amdgcn_wmma_f32_16x16x32_bf16(false, a0c, false, b1c, (short)0, acc[1], false, false);
    acc[5] = __builtin_amdgcn_wmma_f32_16x16x32_bf16(false, a1c, false, b1c, (short)0, acc[5], false, false);
    acc[2] = __builtin_amdgcn_wmma_f32_16x16x32_bf16(false, a0c, false, b2c, (short)0, acc[2], false, false);
    acc[6] = __builtin_amdgcn_wmma_f32_16x16x32_bf16(false, a1c, false, b2c, (short)0, acc[6], false, false);
    acc[3] = __builtin_amdgcn_wmma_f32_16x16x32_bf16(false, a0c, false, b3c, (short)0, acc[3], false, false);
    acc[7] = __builtin_amdgcn_wmma_f32_16x16x32_bf16(false, a1c, false, b3c, (short)0, acc[7], false, false);

    // ---- epilogue: bias / relu / store f32 and/or bf16 ----
    #pragma unroll
    for (int half = 0; half < 2; ++half) {
        #pragma unroll
        for (int j = 0; j < 4; ++j) {
            const int col = nBase + j * 16 + lrow;
            const float bv = bias ? bias[col] : 0.0f;
            #pragma unroll
            for (int i = 0; i < 8; ++i) {
                const int m = mBase + half * 16 + hsel * 8 + i;
                float v = acc[half * 4 + j][i] + bv;
                if (relu) v = v > 0.0f ? v : 0.0f;
                const size_t idx = (size_t)m * N + col;
                if (outF) outF[idx] = v;
                if (outB) outB[idx] = f32_to_bf16(v);
            }
        }
    }
}

// ---------------- elementwise / fusion kernels ----------------

__global__ void init_kernel(const float* __restrict__ prev_belief,
                            const float* __restrict__ prev_state,
                            float* __restrict__ beliefF,
                            unsigned short* __restrict__ beliefB,
                            float* __restrict__ stateF) {
    int i = blockIdx.x * blockDim.x + threadIdx.x;
    if (i < Bn * BEL) {
        float v = prev_belief[i];
        beliefF[i] = v;
        beliefB[i] = f32_to_bf16(v);
    }
    if (i < Bn * ST) stateF[i] = prev_state[i];
}

// sa_in[b, 0:ST] = state*nt ; sa_in[b, ST:ST+ACT] = action  (bf16)
__global__ void build_sa_kernel(const float* __restrict__ stateF,
                                const float* __restrict__ act_t,
                                const float* __restrict__ nt_t, // nullable (t==0)
                                unsigned short* __restrict__ saB) {
    int i = blockIdx.x * blockDim.x + threadIdx.x;
    if (i >= Bn * (ST + ACT)) return;
    int b = i / (ST + ACT);
    int c = i - b * (ST + ACT);
    float v;
    if (c < ST) {
        float n = nt_t ? nt_t[b] : 1.0f;
        v = stateF[b * ST + c] * n;
    } else {
        v = act_t[b * ACT + (c - ST)];
    }
    saB[i] = f32_to_bf16(v);
}

// GRU gating + belief update; emits f32 output slab, bf16 copy, and the
// belief half of the posterior-concat input.
__global__ void gru_kernel(const float* __restrict__ gi,
                           const float* __restrict__ gh,
                           const float* __restrict__ b_hn,
                           float* __restrict__ beliefF,
                           unsigned short* __restrict__ beliefB,
                           unsigned short* __restrict__ hqIn,
                           float* __restrict__ outBelief) {
    int i = blockIdx.x * blockDim.x + threadIdx.x;
    if (i >= Bn * BEL) return;
    int b = i / BEL;
    int j = i - b * BEL;
    const float* giR = gi + (size_t)b * 3 * BEL;
    const float* ghR = gh + (size_t)b * 3 * BEL;
    float r = sigmoidf_(giR[j] + ghR[j]);
    float z = sigmoidf_(giR[BEL + j] + ghR[BEL + j]);
    float n = tanhf(giR[2 * BEL + j] + r * (ghR[2 * BEL + j] + b_hn[j]));
    float bel = (1.0f - z) * n + z * beliefF[i];
    beliefF[i] = bel;
    unsigned short bb = f32_to_bf16(bel);
    beliefB[i] = bb;
    hqIn[(size_t)b * (BEL + EMB) + j] = bb;
    outBelief[i] = bel;
}

__global__ void pack_obs_kernel(const float* __restrict__ obs_t,
                                unsigned short* __restrict__ hqIn) {
    int i = blockIdx.x * blockDim.x + threadIdx.x;
    if (i >= Bn * EMB) return;
    int b = i / EMB;
    int e = i - b * EMB;
    hqIn[(size_t)b * (BEL + EMB) + BEL + e] = f32_to_bf16(obs_t[i]);
}

// split mean/std, softplus + MIN_STD, reparameterize
__global__ void head_kernel(const float* __restrict__ raw,     // [B, 2*ST]
                            const float* __restrict__ noise_t, // [B, ST]
                            float* __restrict__ outMean,
                            float* __restrict__ outStd,
                            float* __restrict__ outState,
                            float* __restrict__ stateNext) {   // nullable
    int i = blockIdx.x * blockDim.x + threadIdx.x;
    if (i >= Bn * ST) return;
    int b = i / ST;
    int s = i - b * ST;
    float m  = raw[(size_t)b * 2 * ST + s];
    float sr = raw[(size_t)b * 2 * ST + ST + s];
    float sp = (sr > 20.0f) ? sr : log1pf(__expf(sr));
    float sd = sp + 0.1f;
    float st = m + sd * noise_t[i];
    outMean[i]  = m;
    outStd[i]   = sd;
    outState[i] = st;
    if (stateNext) stateNext[i] = st;
}

// ---------------- host-side orchestration ----------------

extern "C" void kernel_launch(void* const* d_in, const int* in_sizes, int n_in,
                              void* d_out, int out_size, void* d_ws, size_t ws_size,
                              hipStream_t stream) {
    (void)in_sizes; (void)n_in; (void)out_size; (void)ws_size;

    const float* prev_state   = (const float*)d_in[0];
    const float* actions      = (const float*)d_in[1];
    const float* prev_belief  = (const float*)d_in[2];
    const float* observations = (const float*)d_in[3];
    const float* nonterminals = (const float*)d_in[4];
    const float* prior_noise  = (const float*)d_in[5];
    const float* post_noise   = (const float*)d_in[6];
    const float* W_sa = (const float*)d_in[7];  const float* b_sa = (const float*)d_in[8];
    const float* W_ix = (const float*)d_in[9];  const float* b_ix = (const float*)d_in[10];
    const float* W_hx = (const float*)d_in[11]; const float* b_hn = (const float*)d_in[12];
    const float* W_p1 = (const float*)d_in[13]; const float* b_p1 = (const float*)d_in[14];
    const float* W_p2 = (const float*)d_in[15]; const float* b_p2 = (const float*)d_in[16];
    const float* W_q1 = (const float*)d_in[17]; const float* b_q1 = (const float*)d_in[18];
    const float* W_q2 = (const float*)d_in[19]; const float* b_q2 = (const float*)d_in[20];

    // output slabs (reference return order)
    float* out          = (float*)d_out;
    float* outBeliefs   = out;
    float* outPriorSt   = outBeliefs + (size_t)Tn * Bn * BEL;
    float* outPriorMean = outPriorSt + (size_t)Tn * Bn * ST;
    float* outPriorStd  = outPriorMean + (size_t)Tn * Bn * ST;
    float* outPostSt    = outPriorStd + (size_t)Tn * Bn * ST;
    float* outPostMean  = outPostSt + (size_t)Tn * Bn * ST;
    float* outPostStd   = outPostMean + (size_t)Tn * Bn * ST;

    // workspace carve-out
    char* ws = (char*)d_ws;
    size_t off = 0;
    auto carve = [&](size_t bytes) -> void* {
        void* p = ws + off;
        off += (bytes + 255) & ~(size_t)255;
        return p;
    };
    auto carveH = [&](size_t elems) { return (unsigned short*)carve(elems * 2); };
    auto carveF = [&](size_t elems) { return (float*)carve(elems * 4); };

    unsigned short* WsaP = carveH((size_t)(ST + ACT) * BEL);
    unsigned short* WixP = carveH((size_t)BEL * 3 * BEL);
    unsigned short* WhxP = carveH((size_t)BEL * 3 * BEL);
    unsigned short* Wp1P = carveH((size_t)BEL * HID);
    unsigned short* Wp2P = carveH((size_t)HID * 2 * ST);
    unsigned short* Wq1P = carveH((size_t)(BEL + EMB) * HID);
    unsigned short* Wq2P = carveH((size_t)HID * 2 * ST);

    float*          beliefF = carveF((size_t)Bn * BEL);
    unsigned short* beliefB = carveH((size_t)Bn * BEL);
    float*          stateF  = carveF((size_t)Bn * ST);
    unsigned short* saB     = carveH((size_t)Bn * (ST + ACT));
    unsigned short* hB      = carveH((size_t)Bn * BEL);
    float*          giF     = carveF((size_t)Bn * 3 * BEL);
    float*          ghF     = carveF((size_t)Bn * 3 * BEL);
    unsigned short* hqIn    = carveH((size_t)Bn * (BEL + EMB));
    unsigned short* hpB     = carveH((size_t)Bn * HID);
    unsigned short* hqB     = carveH((size_t)Bn * HID);
    float*          rawP    = carveF((size_t)Bn * 2 * ST);
    float*          rawQ    = carveF((size_t)Bn * 2 * ST);

    // ---- one-time (per call) weight repack to bf16 WMMA fragments ----
    auto pack = [&](const float* W, unsigned short* Wp, int K, int N) {
        size_t total = (size_t)K * N;
        int blocks = (int)((total + 255) / 256);
        pack_w_kernel<<<dim3(blocks), dim3(256), 0, stream>>>(W, Wp, K, N);
    };
    pack(W_sa, WsaP, ST + ACT, BEL);
    pack(W_ix, WixP, BEL, 3 * BEL);
    pack(W_hx, WhxP, BEL, 3 * BEL);
    pack(W_p1, Wp1P, BEL, HID);
    pack(W_p2, Wp2P, HID, 2 * ST);
    pack(W_q1, Wq1P, BEL + EMB, HID);
    pack(W_q2, Wq2P, HID, 2 * ST);

    init_kernel<<<dim3((Bn * BEL + 255) / 256), dim3(256), 0, stream>>>(
        prev_belief, prev_state, beliefF, beliefB, stateF);

    auto gemm = [&](const unsigned short* A, const unsigned short* Bp,
                    const float* bias, float* oF, unsigned short* oB,
                    int K, int N, int relu) {
        dim3 grid(Bn / 32, N / 256);
        gemm_wmma_bf16<<<grid, dim3(128), 0, stream>>>(A, Bp, bias, oF, oB, K, N, relu);
    };

    const int gBEL = (Bn * BEL + 255) / 256;
    const int gSA  = (Bn * (ST + ACT) + 255) / 256;
    const int gOBS = (Bn * EMB + 255) / 256;
    const int gST  = (Bn * ST + 255) / 256;

    for (int t = 0; t < Tn; ++t) {
        const float* act_t = actions + (size_t)t * Bn * ACT;
        const float* obs_t = observations + (size_t)t * Bn * EMB;
        const float* nt_t  = (t == 0) ? nullptr : nonterminals + (size_t)(t - 1) * Bn;

        // h = relu([state*nt, action] @ W_sa + b_sa)  -> bf16
        build_sa_kernel<<<dim3(gSA), dim3(256), 0, stream>>>(stateF, act_t, nt_t, saB);
        gemm(saB, WsaP, b_sa, nullptr, hB, ST + ACT, BEL, 1);

        // GRU gate pre-activations
        gemm(hB, WixP, b_ix, giF, nullptr, BEL, 3 * BEL, 0);         // gi
        gemm(beliefB, WhxP, nullptr, ghF, nullptr, BEL, 3 * BEL, 0); // gh (old belief)

        // obs half of posterior concat (independent of GRU)
        pack_obs_kernel<<<dim3(gOBS), dim3(256), 0, stream>>>(obs_t, hqIn);

        // belief update (+ bf16 copy + concat fill + output slab)
        gru_kernel<<<dim3(gBEL), dim3(256), 0, stream>>>(
            giF, ghF, b_hn, beliefF, beliefB, hqIn,
            outBeliefs + (size_t)t * Bn * BEL);

        // prior head
        gemm(beliefB, Wp1P, b_p1, nullptr, hpB, BEL, HID, 1);
        gemm(hpB, Wp2P, b_p2, rawP, nullptr, HID, 2 * ST, 0);
        head_kernel<<<dim3(gST), dim3(256), 0, stream>>>(
            rawP, prior_noise + (size_t)t * Bn * ST,
            outPriorMean + (size_t)t * Bn * ST,
            outPriorStd + (size_t)t * Bn * ST,
            outPriorSt + (size_t)t * Bn * ST, nullptr);

        // posterior head (also produces next step's state)
        gemm(hqIn, Wq1P, b_q1, nullptr, hqB, BEL + EMB, HID, 1);
        gemm(hqB, Wq2P, b_q2, rawQ, nullptr, HID, 2 * ST, 0);
        head_kernel<<<dim3(gST), dim3(256), 0, stream>>>(
            rawQ, post_noise + (size_t)t * Bn * ST,
            outPostMean + (size_t)t * Bn * ST,
            outPostStd + (size_t)t * Bn * ST,
            outPostSt + (size_t)t * Bn * ST, stateF);
    }
}
